// MambaModel_88914412962409
// MI455X (gfx1250) — compile-verified, matrix operand order checked by reading, not run on hardware
//
#include <hip/hip_runtime.h>
#include <hip/hip_bf16.h>
#include <stdint.h>

// ---------------------------------------------------------------------------
// Mamba forward for MI455X (gfx1250, wave32).
// GEMMs: bf16 WMMA (v_wmma_f32_16x16x32_bf16), f32 accumulate. The B operand
// tile is staged into LDS by the Tensor Data Mover (tensor_load_to_lds +
// s_wait_tensorcnt), double-buffered so the DMA overlaps the WMMA work.
// Selective scan runs fp32, one (b,e,n) state per lane, 16-lane shfl_xor
// reduction (wave32-native).
// ---------------------------------------------------------------------------

#define B_SZ     4
#define SEQ      2048
#define D_MODEL  384
#define D_INNER  768
#define D_STATE  16
#define DT_RANK  24
#define VOCAB    5000
#define N_LAYER  4
#define ROWS     (B_SZ * SEQ)          // 8192 token rows
#define EPS      1e-5f

#define TDM_BN   128                   // block N tile staged by TDM
#define TDM_BK   32                    // K step (one WMMA K)

typedef __attribute__((ext_vector_type(16))) __bf16 v16bf;
typedef __attribute__((ext_vector_type(8)))  float  v8f;
typedef __attribute__((ext_vector_type(4)))  unsigned int u32x4;
typedef __attribute__((ext_vector_type(8)))  int i32x8;
typedef __attribute__((ext_vector_type(4)))  int i32x4;

union Frag {
    v16bf v;
    uint32_t u[8];
};

__device__ __forceinline__ unsigned short f32_to_bf16(float f) {
    unsigned int u = __builtin_bit_cast(unsigned int, f);
    unsigned int r = u + 0x7FFFu + ((u >> 16) & 1u);   // round-to-nearest-even
    return (unsigned short)(r >> 16);
}

__device__ __forceinline__ float silu_f(float x) {
    return x / (1.f + __expf(-x));
}

// 16x32 bf16 WMMA operand fragment from global. Per ISA 16-bit operand layout:
// lanes 0-15 hold K chunks {0..7, 16..23}, lanes 16-31 hold {8..15, 24..31}.
__device__ __forceinline__ void load_frag(Frag& f, const unsigned short* __restrict__ base,
                                          int ld, int row, int k, int khalf) {
    const uint4* p0 = reinterpret_cast<const uint4*>(base + (size_t)row * ld + k + khalf * 8);
    const uint4* p1 = reinterpret_cast<const uint4*>(base + (size_t)row * ld + k + 16 + khalf * 8);
    uint4 a = *p0, b = *p1;
    f.u[0] = a.x; f.u[1] = a.y; f.u[2] = a.z; f.u[3] = a.w;
    f.u[4] = b.x; f.u[5] = b.y; f.u[6] = b.z; f.u[7] = b.w;
}

// Same fragment shape out of the LDS-staged B tile (row-major [TDM_BN][TDM_BK]).
__device__ __forceinline__ void load_frag_lds(Frag& f, const unsigned short* base,
                                              int row, int khalf) {
    const uint4* p0 = reinterpret_cast<const uint4*>(base + row * TDM_BK + khalf * 8);
    const uint4* p1 = reinterpret_cast<const uint4*>(base + row * TDM_BK + 16 + khalf * 8);
    uint4 a = *p0, b = *p1;
    f.u[0] = a.x; f.u[1] = a.y; f.u[2] = a.z; f.u[3] = a.w;
    f.u[4] = b.x; f.u[5] = b.y; f.u[6] = b.z; f.u[7] = b.w;
}

// C = A(MxK bf16) * B(NxK bf16)^T, f32 out. Block = 8 waves -> 64x128 tile,
// wave -> 32x32 (4 WMMA / K-step). B tile staged to LDS by the TDM, double
// buffered. Requires M%64==0, K%32==0, Npad%128==0; stores guarded at Nstore.
__global__ void gemm_bf16_wmma(const unsigned short* __restrict__ A,
                               const unsigned short* __restrict__ B,
                               float* __restrict__ C,
                               int M, int Npad, int Nstore, int K, int ldc) {
    __shared__ unsigned short ldsB[2][TDM_BN * TDM_BK];   // 2 x 8 KB

    const int lane    = threadIdx.x & 31;
    const int wave    = threadIdx.x >> 5;
    const int wm      = wave >> 2;                 // 0..1
    const int wn      = wave & 3;                  // 0..3
    const int m_tile  = blockIdx.y * 64 + wm * 32;
    const int n_tile0 = blockIdx.x * TDM_BN;       // block N origin
    const int n_local = wn * 32;
    const int r       = lane & 15;
    const int khalf   = lane >> 4;

    // Wave 0 issues one TDM descriptor: 2D tile (32 K) x (128 N-rows) of bf16
    // from B[n_tile0 + :, k + :] into ldsB[buf]. D# packing per ISA ch.8.
    auto tdm_issue = [&](int k, int buf) {
        if (threadIdx.x < 32) {
            unsigned long long ga = (unsigned long long)(uintptr_t)(B + (size_t)n_tile0 * K + k);
            unsigned int la = (unsigned int)(uintptr_t)(&ldsB[buf][0]);   // low 32 = LDS offset
            u32x4 g0 = {};
            g0[0] = 1u;                                          // count=1 (valid), user mode
            g0[1] = la;                                          // lds_addr
            g0[2] = (unsigned int)(ga & 0xFFFFFFFFu);            // global_addr[31:0]
            g0[3] = (unsigned int)((ga >> 32) & 0x01FFFFFFu)     // global_addr[56:32]
                    | (2u << 30);                                // type = 2 (image)
            i32x8 g1 = {};
            g1[0] = (int)(1u << 16);                             // data_size=1 -> 2 bytes
            g1[1] = (int)(((unsigned)K & 0xFFFFu) << 16);        // tensor_dim0 lo
            g1[2] = (int)((((unsigned)K >> 16) & 0xFFFFu)
                    | (((unsigned)Npad & 0xFFFFu) << 16));       // dim0 hi | tensor_dim1 lo
            g1[3] = (int)((((unsigned)Npad >> 16) & 0xFFFFu)
                    | ((unsigned)TDM_BK << 16));                 // dim1 hi | tile_dim0=32
            g1[4] = (int)TDM_BN;                                 // tile_dim1=128, tile_dim2=0
            g1[5] = (int)(unsigned)K;                            // tensor_dim0_stride lo
            g1[6] = 0;                                           // stride hi | dim1_stride lo
            g1[7] = 0;
            i32x4 g2 = {}, g3 = {};
            i32x8 g4 = {};                                       // unused trailing group
            __builtin_amdgcn_tensor_load_to_lds(g0, g1, g2, g3, g4, 0);
        }
    };
    auto tdm_wait = [&]() {
        if (threadIdx.x < 32) __builtin_amdgcn_s_wait_tensorcnt(0);
    };

    v8f c00 = {}, c01 = {}, c10 = {}, c11 = {};

    int buf = 0;
    tdm_issue(0, 0);
    tdm_wait();
    __syncthreads();

    for (int k = 0; k < K; k += 32) {
        if (k + 32 < K) {
            tdm_issue(k + 32, buf ^ 1);              // TDM prefetch of next B tile
            __builtin_prefetch(A + (size_t)(m_tile + r) * K + (k + 32) + khalf * 8, 0, 3);
            __builtin_prefetch(A + (size_t)(m_tile + 16 + r) * K + (k + 32) + khalf * 8, 0, 3);
        }
        Frag a0, a1, b0, b1;
        load_frag(a0, A, K, m_tile + r,      k, khalf);
        load_frag(a1, A, K, m_tile + 16 + r, k, khalf);
        load_frag_lds(b0, &ldsB[buf][0], n_local + r,      khalf);
        load_frag_lds(b1, &ldsB[buf][0], n_local + 16 + r, khalf);

        c00 = __builtin_amdgcn_wmma_f32_16x16x32_bf16(false, a0.v, false, b0.v, (short)0, c00, false, false);
        c01 = __builtin_amdgcn_wmma_f32_16x16x32_bf16(false, a0.v, false, b1.v, (short)0, c01, false, false);
        c10 = __builtin_amdgcn_wmma_f32_16x16x32_bf16(false, a1.v, false, b0.v, (short)0, c10, false, false);
        c11 = __builtin_amdgcn_wmma_f32_16x16x32_bf16(false, a1.v, false, b1.v, (short)0, c11, false, false);

        tdm_wait();            // wave0: next tile landed
        __syncthreads();       // publish buf^1, retire reads of buf
        buf ^= 1;
    }

    // C/D layout: lane -> col = n&15; VGPR i -> row = i + 8*(lane>>4)
    const int col0 = n_tile0 + n_local + r;
    const int col1 = n_tile0 + n_local + 16 + r;
    const int row0 = m_tile + khalf * 8;
    const int row1 = m_tile + 16 + khalf * 8;
    #pragma unroll
    for (int i = 0; i < 8; ++i) {
        if (col0 < Nstore) {
            C[(size_t)(row0 + i) * ldc + col0] = c00[i];
            C[(size_t)(row1 + i) * ldc + col0] = c10[i];
        }
        if (col1 < Nstore) {
            C[(size_t)(row0 + i) * ldc + col1] = c01[i];
            C[(size_t)(row1 + i) * ldc + col1] = c11[i];
        }
    }
}

// x[row] = embed[tokens[row]]
__global__ void embed_kernel(const int* __restrict__ tokens, const float* __restrict__ embed,
                             float* __restrict__ x) {
    int idx = blockIdx.x * blockDim.x + threadIdx.x;
    if (idx >= ROWS * D_MODEL) return;
    int row = idx / D_MODEL, c = idx - row * D_MODEL;
    x[idx] = embed[(size_t)tokens[row] * D_MODEL + c];
}

// fp32 -> bf16 with zero padding (weights / embed staging for WMMA B operand)
__global__ void convert_pad_kernel(const float* __restrict__ src, unsigned short* __restrict__ dst,
                                   int rows, int cols, int rowsPad, int colsPad) {
    int idx = blockIdx.x * blockDim.x + threadIdx.x;
    if (idx >= rowsPad * colsPad) return;
    int r = idx / colsPad, c = idx - r * colsPad;
    float v = (r < rows && c < cols) ? src[(size_t)r * cols + c] : 0.f;
    dst[idx] = f32_to_bf16(v);
}

// RMSNorm, one wave per row, bf16 output feeds the next GEMM's A operand.
__global__ void rmsnorm_bf16_kernel(const float* __restrict__ x, const float* __restrict__ w,
                                    unsigned short* __restrict__ out) {
    int wave = (blockIdx.x * blockDim.x + threadIdx.x) >> 5;
    int lane = threadIdx.x & 31;
    if (wave >= ROWS) return;
    const float* xr = x + (size_t)wave * D_MODEL;
    float ss = 0.f;
    for (int i = lane; i < D_MODEL; i += 32) { float v = xr[i]; ss += v * v; }
    #pragma unroll
    for (int off = 16; off; off >>= 1) ss += __shfl_xor(ss, off, 32);
    float inv = rsqrtf(ss / (float)D_MODEL + EPS);
    unsigned short* orow = out + (size_t)wave * D_MODEL;
    for (int i = lane; i < D_MODEL; i += 32) orow[i] = f32_to_bf16(xr[i] * inv * w[i]);
}

// causal depthwise conv (k=4, left pad 3) + bias + silu; f32 (scan) and bf16 (x_proj A)
__global__ void conv_silu_kernel(const float* __restrict__ xz, const float* __restrict__ cw,
                                 const float* __restrict__ cb, float* __restrict__ xin_f32,
                                 unsigned short* __restrict__ xin_bf16) {
    int idx = blockIdx.x * blockDim.x + threadIdx.x;
    if (idx >= ROWS * D_INNER) return;
    int e  = idx % D_INNER;
    int bl = idx / D_INNER;
    int l  = bl % SEQ;
    int b  = bl / SEQ;
    float acc = cb[e];
    #pragma unroll
    for (int j = 0; j < 4; ++j) {
        int ls = l - 3 + j;
        if (ls >= 0) acc += cw[e * 4 + j] * xz[(size_t)(b * SEQ + ls) * (2 * D_INNER) + e];
    }
    float s = silu_f(acc);
    xin_f32[idx]  = s;
    xin_bf16[idx] = f32_to_bf16(s);
}

// proj[:, :24] -> zero-padded 8192x32 bf16 A operand for dt_proj GEMM (K pad 24->32)
__global__ void dt_stage_kernel(const float* __restrict__ proj, unsigned short* __restrict__ dtA) {
    int idx = blockIdx.x * blockDim.x + threadIdx.x;
    if (idx >= ROWS * 32) return;
    int r = idx >> 5, c = idx & 31;
    float v = (c < DT_RANK) ? proj[(size_t)r * 64 + c] : 0.f;
    dtA[idx] = f32_to_bf16(v);
}

// dt = softplus(dt_raw + bias), in place
__global__ void softplus_kernel(float* __restrict__ dt, const float* __restrict__ bias) {
    int idx = blockIdx.x * blockDim.x + threadIdx.x;
    if (idx >= ROWS * D_INNER) return;
    float v = dt[idx] + bias[idx % D_INNER];
    dt[idx] = (v > 20.f) ? v : log1pf(__expf(v));
}

// Selective scan: one (b,e,n) state per lane; 16 lanes per channel.
// h[n] = h[n]*exp(dt*A[e,n]) + dt*x*B[n];  y = sum_n h[n]*C[n]
__global__ void scan_kernel(const float* __restrict__ dt, const float* __restrict__ xin,
                            const float* __restrict__ proj, const float* __restrict__ A_log,
                            float* __restrict__ y_scan) {
    int gid = blockIdx.x * blockDim.x + threadIdx.x;
    int ch  = gid >> 4;
    int n   = gid & 15;
    if (ch >= B_SZ * D_INNER) return;
    int b = ch / D_INNER;
    int e = ch - b * D_INNER;
    float Aen = -__expf(A_log[(size_t)e * D_STATE + n]);
    float h = 0.f;
    size_t base = (size_t)b * SEQ;
    for (int l = 0; l < SEQ; ++l) {
        size_t row = base + l;
        float dt_v = dt [row * D_INNER + e];
        float x_v  = xin[row * D_INNER + e];
        float bn   = proj[row * 64 + DT_RANK + n];
        float cn   = proj[row * 64 + DT_RANK + D_STATE + n];
        h = h * __expf(dt_v * Aen) + dt_v * x_v * bn;
        float p = h * cn;
        p += __shfl_xor(p, 1, 16);
        p += __shfl_xor(p, 2, 16);
        p += __shfl_xor(p, 4, 16);
        p += __shfl_xor(p, 8, 16);
        if (n == 0) y_scan[row * D_INNER + e] = p;
    }
}

// y = (y_scan + xin*D) * silu(z), bf16 out (A operand of out_proj)
__global__ void gate_kernel(const float* __restrict__ y_scan, const float* __restrict__ xin,
                            const float* __restrict__ xz, const float* __restrict__ Dp,
                            unsigned short* __restrict__ ybf) {
    int idx = blockIdx.x * blockDim.x + threadIdx.x;
    if (idx >= ROWS * D_INNER) return;
    int e = idx % D_INNER, row = idx / D_INNER;
    float z = xz[(size_t)row * (2 * D_INNER) + D_INNER + e];
    float y = (y_scan[idx] + xin[idx] * Dp[e]) * silu_f(z);
    ybf[idx] = f32_to_bf16(y);
}

__global__ void residual_kernel(float* __restrict__ x, const float* __restrict__ y) {
    int idx = blockIdx.x * blockDim.x + threadIdx.x;
    if (idx >= ROWS * D_MODEL) return;
    x[idx] += y[idx];
}

// Npad must be a multiple of 128 (B buffer zero-padded to Npad rows).
static inline void launch_gemm(const unsigned short* A, const unsigned short* B, float* C,
                               int M, int Npad, int Nstore, int K, int ldc, hipStream_t s) {
    dim3 grid(Npad / TDM_BN, M / 64);
    gemm_bf16_wmma<<<grid, dim3(256), 0, s>>>(A, B, C, M, Npad, Nstore, K, ldc);
}

static inline dim3 g1(int n) { return dim3((n + 255) / 256); }

extern "C" void kernel_launch(void* const* d_in, const int* in_sizes, int n_in,
                              void* d_out, int out_size, void* d_ws, size_t ws_size,
                              hipStream_t stream) {
    const int*   tokens  = (const int*)  d_in[0];
    const float* embed   = (const float*)d_in[1];
    const float* norm_w  = (const float*)d_in[2];
    const float* in_w    = (const float*)d_in[3];
    const float* conv_w  = (const float*)d_in[4];
    const float* conv_b  = (const float*)d_in[5];
    const float* xp_w    = (const float*)d_in[6];
    const float* dt_w    = (const float*)d_in[7];
    const float* dt_b    = (const float*)d_in[8];
    const float* A_log   = (const float*)d_in[9];
    const float* D_param = (const float*)d_in[10];
    const float* out_w   = (const float*)d_in[11];
    const float* fnorm_w = (const float*)d_in[12];
    float* logits = (float*)d_out;

    char* ws = (char*)d_ws;
    size_t off = 0;
    auto alloc = [&](size_t bytes) -> void* {
        void* p = ws + off;
        off = (off + bytes + 255) & ~(size_t)255;
        return p;
    };
    float*          x_f32    = (float*)          alloc((size_t)ROWS * D_MODEL * 4);
    unsigned short* xn_bf    = (unsigned short*) alloc((size_t)ROWS * D_MODEL * 2);
    float*          xz_f32   = (float*)          alloc((size_t)ROWS * 2 * D_INNER * 4);
    float*          xin_f32  = (float*)          alloc((size_t)ROWS * D_INNER * 4);
    unsigned short* xin_bf   = (unsigned short*) alloc((size_t)ROWS * D_INNER * 2);
    float*          proj_pad = (float*)          alloc((size_t)ROWS * 64 * 4);
    unsigned short* dtA_bf   = (unsigned short*) alloc((size_t)ROWS * 32 * 2);
    float*          dt_f32   = (float*)          alloc((size_t)ROWS * D_INNER * 4);
    float*          yscan    = (float*)          alloc((size_t)ROWS * D_INNER * 4);
    unsigned short* y_bf     = (unsigned short*) alloc((size_t)ROWS * D_INNER * 2);
    float*          ycore    = (float*)          alloc((size_t)ROWS * D_MODEL * 4);
    unsigned short* w_in_bf  = (unsigned short*) alloc((size_t)2 * D_INNER * D_MODEL * 2);
    unsigned short* w_xp_bf  = (unsigned short*) alloc((size_t)128 * D_INNER * 2);   // 56 -> 128
    unsigned short* w_dt_bf  = (unsigned short*) alloc((size_t)D_INNER * 32 * 2);
    unsigned short* w_out_bf = (unsigned short*) alloc((size_t)D_MODEL * D_INNER * 2);
    unsigned short* emb_bf   = (unsigned short*) alloc((size_t)5120 * D_MODEL * 2);  // 5000 -> 5120

    // embedding gather + bf16 embed staging (LM head B operand, N pad 5000->5120)
    embed_kernel<<<g1(ROWS * D_MODEL), 256, 0, stream>>>(tokens, embed, x_f32);
    convert_pad_kernel<<<g1(5120 * D_MODEL), 256, 0, stream>>>(embed, emb_bf, VOCAB, D_MODEL, 5120, D_MODEL);

    for (int i = 0; i < N_LAYER; ++i) {
        const float* nw  = norm_w  + (size_t)i * D_MODEL;
        const float* iw  = in_w    + (size_t)i * 2 * D_INNER * D_MODEL;
        const float* cwi = conv_w  + (size_t)i * D_INNER * 4;
        const float* cbi = conv_b  + (size_t)i * D_INNER;
        const float* xpi = xp_w    + (size_t)i * (DT_RANK + 2 * D_STATE) * D_INNER;
        const float* dwi = dt_w    + (size_t)i * D_INNER * DT_RANK;
        const float* dbi = dt_b    + (size_t)i * D_INNER;
        const float* Ali = A_log   + (size_t)i * D_INNER * D_STATE;
        const float* Dpi = D_param + (size_t)i * D_INNER;
        const float* owi = out_w   + (size_t)i * D_MODEL * D_INNER;

        // rmsnorm -> bf16 A; in_proj GEMM (8192 x 384 -> 1536)
        rmsnorm_bf16_kernel<<<dim3(ROWS / 8), 256, 0, stream>>>(x_f32, nw, xn_bf);
        convert_pad_kernel<<<g1(2 * D_INNER * D_MODEL), 256, 0, stream>>>(
            iw, w_in_bf, 2 * D_INNER, D_MODEL, 2 * D_INNER, D_MODEL);
        launch_gemm(xn_bf, w_in_bf, xz_f32, ROWS, 2 * D_INNER, 2 * D_INNER, D_MODEL, 2 * D_INNER, stream);

        // depthwise conv + silu
        conv_silu_kernel<<<g1(ROWS * D_INNER), 256, 0, stream>>>(xz_f32, cwi, cbi, xin_f32, xin_bf);

        // x_proj GEMM (8192 x 768 -> 56, N padded to 128; store into 64-stride buffer)
        convert_pad_kernel<<<g1(128 * D_INNER), 256, 0, stream>>>(
            xpi, w_xp_bf, DT_RANK + 2 * D_STATE, D_INNER, 128, D_INNER);
        launch_gemm(xin_bf, w_xp_bf, proj_pad, ROWS, 128, 64, D_INNER, 64, stream);

        // dt_proj GEMM (8192 x 24(->32) -> 768) + softplus(bias)
        dt_stage_kernel<<<g1(ROWS * 32), 256, 0, stream>>>(proj_pad, dtA_bf);
        convert_pad_kernel<<<g1(D_INNER * 32), 256, 0, stream>>>(dwi, w_dt_bf, D_INNER, DT_RANK, D_INNER, 32);
        launch_gemm(dtA_bf, w_dt_bf, dt_f32, ROWS, D_INNER, D_INNER, 32, D_INNER, stream);
        softplus_kernel<<<g1(ROWS * D_INNER), 256, 0, stream>>>(dt_f32, dbi);

        // selective scan: 4*768 channels x 16 state lanes = 49152 threads
        scan_kernel<<<dim3((B_SZ * D_INNER * 16) / 256), 256, 0, stream>>>(
            dt_f32, xin_f32, proj_pad, Ali, yscan);

        // D skip + gate, out_proj GEMM, residual
        gate_kernel<<<g1(ROWS * D_INNER), 256, 0, stream>>>(yscan, xin_f32, xz_f32, Dpi, y_bf);
        convert_pad_kernel<<<g1(D_MODEL * D_INNER), 256, 0, stream>>>(
            owi, w_out_bf, D_MODEL, D_INNER, D_MODEL, D_INNER);
        launch_gemm(y_bf, w_out_bf, ycore, ROWS, D_MODEL, D_MODEL, D_INNER, D_MODEL, stream);
        residual_kernel<<<g1(ROWS * D_MODEL), 256, 0, stream>>>(x_f32, ycore);
    }

    // final rmsnorm + tied LM head (8192 x 384 -> 5000, N padded 5120, stores guarded)
    rmsnorm_bf16_kernel<<<dim3(ROWS / 8), 256, 0, stream>>>(x_f32, fnorm_w, xn_bf);
    launch_gemm(xn_bf, emb_bf, logits, ROWS, 5120, VOCAB, D_MODEL, VOCAB, stream);
}
